// SGC_B_50448685859076
// MI455X (gfx1250) — compile-verified, matrix operand order checked by reading, not run on hardware
//
#include <hip/hip_runtime.h>
#include <hip/hip_bf16.h>

// CDNA5 / gfx1250: wave32, WMMA bf16 16x16x32, f32 accumulate.
typedef __attribute__((ext_vector_type(16))) __bf16 v16bf;
typedef __attribute__((ext_vector_type(8)))  float  v8f;
typedef __attribute__((ext_vector_type(4)))  float  f4;   // native vector, AS-cast friendly

#define HDIM 128   // D == H == 128 in this problem

// Explicit global-AS load type so staging lowers to global_load_b128 (LOADcnt
// only) instead of flat_load_b128 (LOADcnt+DScnt). Use the native vector type:
// HIP's float4 is a class and cannot copy from an AS(1) reference.
typedef const __attribute__((address_space(1))) f4* as1_cf4;

// Async global->LDS path (gfx1250). Param 0 is `int4 __device__*` per hipcc.
#if defined(__AMDGCN__) && __has_builtin(__builtin_amdgcn_global_load_async_to_lds_b128) && __has_builtin(__builtin_amdgcn_s_wait_asynccnt)
#define USE_ASYNC_LDS 1
typedef int v4i __attribute__((vector_size(16)));
typedef __attribute__((address_space(1))) v4i* as1_v4;
typedef __attribute__((address_space(3))) v4i* as3_v4;
#else
#define USE_ASYNC_LDS 0
#endif

union Frag16 { v16bf v; uint4 q[2]; };

// ---------------- degree / norm kernels ----------------

__global__ void k_deg(const int* __restrict__ dst, float* __restrict__ deg, int E) {
    int e = blockIdx.x * blockDim.x + threadIdx.x;
    if (e < E) atomicAdd(&deg[dst[e]], 1.0f);
}

__global__ void k_norm(float* __restrict__ norm, float* __restrict__ norm2, int N) {
    int i = blockIdx.x * blockDim.x + threadIdx.x;
    if (i < N) {
        float d = fmaxf(norm2[i], 1.0f);   // deg was accumulated into norm2 slot
        norm[i]  = 1.0f / sqrtf(d);
        norm2[i] = 1.0f / d;
    }
}

// ---------------- weight f32 -> bf16 pre-conversion (once per launch) -------

__global__ void k_cvtW(const float* __restrict__ w, __bf16* __restrict__ o, int n) {
    int i = blockIdx.x * blockDim.x + threadIdx.x;
    if (i < n) o[i] = (__bf16)w[i];
}

// ---------------- SpMM: agg[dst] += h[src] * norm[src] ----------------
// One wave per edge: 32 lanes x float4 = full 128-col row. Gathers are
// L2-resident (h fits in the 192MB L2); scatter is per-dword f32 atomics.

__global__ __launch_bounds__(256) void k_spmm(const float* __restrict__ h,
                                              const float* __restrict__ norm,
                                              const int* __restrict__ src,
                                              const int* __restrict__ dst,
                                              float* __restrict__ agg, int E) {
    int e = blockIdx.x * 8 + (threadIdx.x >> 5);   // wave-uniform edge id
    if (e >= E) return;
    int q = threadIdx.x & 31;
    int s = __builtin_amdgcn_readfirstlane(src[e]);
    int d = __builtin_amdgcn_readfirstlane(dst[e]);
    float sc = norm[s];
    f4 v = *(as1_cf4)(const void*)(h + (size_t)s * HDIM + q * 4);
    float* o = agg + (size_t)d * HDIM + q * 4;
    atomicAdd(o + 0, v.x * sc);
    atomicAdd(o + 1, v.y * sc);
    atomicAdd(o + 2, v.z * sc);
    atomicAdd(o + 3, v.w * sc);
}

// ---------------- fused WMMA GEMM + epilogue ----------------
// out[row,col] = mode0: acc + bias[col]
//               mode1: agg[row,col]*norm[row] + norm2[row]*(acc + bias[col])
// A = concat(A0,A1,A2) row-major f32 (cat never materialized), converted to
// bf16 while staged in LDS. W is pre-converted bf16 [128, Ktot] row-major;
// one 32KB W-segment tile per block is staged via async global->LDS copies.
// Block = 256 thr = 8 waves; block owns rows [16bx,16bx+16), wave w cols [16w,16w+16).

__global__ __launch_bounds__(256) void k_gemm(const float* __restrict__ A0,
                                              const float* __restrict__ A1,
                                              const float* __restrict__ A2,
                                              int nseg,
                                              const __bf16* __restrict__ Wb,
                                              const float* __restrict__ bias,
                                              int Ktot,
                                              const float* __restrict__ agg,
                                              const float* __restrict__ norm,
                                              const float* __restrict__ norm2,
                                              float* __restrict__ out,
                                              int mode, int N) {
    __shared__ __bf16 Asb[16 * HDIM];          //  4 KB: A tile, 16 rows x 128 k (bf16)
    __shared__ __bf16 Wsb[HDIM * HDIM];        // 32 KB: W tile, 128 cols x 128 k (bf16)

    const int tid     = threadIdx.x;
    const int lane    = tid & 31;              // wave32
    const int wave    = tid >> 5;              // 0..7
    const int rowBase = blockIdx.x * 16;
    const int colBase = wave * 16;

    const float* segs[3] = { A0, A1, A2 };

    const int m   = lane & 15;                 // row (A) / col (B) within tile
    const int khi = (lane >> 4) & 1;           // lanes 16..31 take the +8 K offset

    v8f c = {};

    for (int s = 0; s < nseg; ++s) {
        const float* Ag = segs[s];
        __syncthreads();                       // protect LDS tiles from prev iter

        // ---- stage A tile: 16 rows x 128 f32 -> bf16 in LDS (cvt once, shared) ----
        // Branch-free: clamp OOB rows; epilogue never stores rows >= N anyway.
        #pragma unroll
        for (int i = 0; i < 2; ++i) {
            int i4  = tid + i * 256;           // 0..511 float4 slots
            int r   = i4 >> 5;                 // 32 float4 per row
            int col = (i4 & 31) * 4;
            int rr  = min(rowBase + r, N - 1);
            f4 v = *(as1_cf4)(const void*)&Ag[(size_t)rr * HDIM + col];
            union { __bf16 h[4]; uint2 u; } pk;
            pk.h[0] = (__bf16)v.x; pk.h[1] = (__bf16)v.y;
            pk.h[2] = (__bf16)v.z; pk.h[3] = (__bf16)v.w;
            *(uint2*)&Asb[r * HDIM + col] = pk.u;
        }

        // ---- stage W segment tile: 128 cols x 128 k bf16 = 2048 x 16B chunks ----
        const int kseg = s * HDIM;
        #pragma unroll
        for (int i = 0; i < 8; ++i) {
            int cidx = tid + i * 256;          // 0..2047 b128 chunks
            int n    = cidx >> 4;              // 16 chunks per W row
            int koff = (cidx & 15) * 8;        // halfs
            const __bf16* gp = Wb + (size_t)n * Ktot + kseg + koff;
            __bf16* lp = &Wsb[n * HDIM + koff];
#if USE_ASYNC_LDS
            __builtin_amdgcn_global_load_async_to_lds_b128(
                (as1_v4)(void*)gp, (as3_v4)(void*)lp, 0, 0);
#else
            *(uint4*)lp = *(const uint4*)gp;
#endif
        }
#if USE_ASYNC_LDS
        __builtin_amdgcn_s_wait_asynccnt(0);
#endif
        if (s + 1 < nseg)                      // warm next A segment
            __builtin_prefetch(&segs[s + 1][(size_t)(rowBase + m) * HDIM + khi * 64], 0, 1);
        __syncthreads();

        // ---- 4 WMMA K-steps: pure ds_load_b128 + v_wmma ----
        #pragma unroll
        for (int k0 = 0; k0 < HDIM; k0 += 32) {
            int ka = k0 + khi * 8;             // ISA 16-bit A/B layout K base for this lane
            Frag16 a, b;
            a.q[0] = *(const uint4*)&Asb[m * HDIM + ka];
            a.q[1] = *(const uint4*)&Asb[m * HDIM + ka + 16];
            b.q[0] = *(const uint4*)&Wsb[(colBase + m) * HDIM + ka];
            b.q[1] = *(const uint4*)&Wsb[(colBase + m) * HDIM + ka + 16];
            c = __builtin_amdgcn_wmma_f32_16x16x32_bf16(
                    false, a.v, false, b.v, (short)0, c, false, false);
        }
    }

    // epilogue: accumulator layout -> VGPR r, lane l: row = r + 8*(l>=16), col = l&15
    const int n = colBase + m;
    float bv = bias[n];
    #pragma unroll
    for (int r = 0; r < 8; ++r) {
        int row = rowBase + r + khi * 8;
        if (row >= N) continue;
        float g = c[r] + bv;
        float o;
        if (mode == 0) {
            o = g;
        } else {
            o = agg[(size_t)row * HDIM + n] * norm[row] + norm2[row] * g;
        }
        out[(size_t)row * HDIM + n] = o;
    }
}

// ---------------- launcher ----------------

extern "C" void kernel_launch(void* const* d_in, const int* in_sizes, int n_in,
                              void* d_out, int out_size, void* d_ws, size_t ws_size,
                              hipStream_t stream) {
    const float* feat = (const float*)d_in[0];
    const int*   src  = (const int*)d_in[1];
    const int*   dst  = (const int*)d_in[2];
    const float* W0   = (const float*)d_in[3];
    const float* b0   = (const float*)d_in[4];
    const float* Wb0  = (const float*)d_in[5];
    const float* bb0  = (const float*)d_in[6];
    const float* Wb1  = (const float*)d_in[7];
    const float* bb1  = (const float*)d_in[8];

    const int N = in_sizes[0] / HDIM;
    const int E = in_sizes[1];

    float* outp  = (float*)d_out;
    float* ws    = (float*)d_ws;
    float* norm  = ws;                               // [N]
    float* norm2 = ws + (size_t)N;                   // [N] (deg accumulated here first)
    float* h0    = norm2 + (size_t)N;                // [N,128]
    float* h1    = h0 + (size_t)N * HDIM;            // [N,128]
    float* agg   = h1 + (size_t)N * HDIM;            // [N,128]
    __bf16* w16  = (__bf16*)(agg + (size_t)N * HDIM);
    __bf16* w0b   = w16;                             // [128*128]
    __bf16* wb0b  = w0b + 128 * HDIM;                // [128*256]
    __bf16* wb1b  = wb0b + 128 * 2 * HDIM;           // [128*384]

    const int mblocks = (N + 15) / 16;
    const int spmmB   = (E + 7) / 8;

    // degrees -> norm, norm2
    (void)hipMemsetAsync(norm2, 0, (size_t)N * sizeof(float), stream);
    k_deg <<<(E + 255) / 256, 256, 0, stream>>>(dst, norm2, E);
    k_norm<<<(N + 255) / 256, 256, 0, stream>>>(norm, norm2, N);

    // weights -> bf16 (once)
    k_cvtW<<<(128 * HDIM + 255) / 256, 256, 0, stream>>>(W0, w0b, 128 * HDIM);
    k_cvtW<<<(128 * 2 * HDIM + 255) / 256, 256, 0, stream>>>(Wb0, wb0b, 128 * 2 * HDIM);
    k_cvtW<<<(128 * 3 * HDIM + 255) / 256, 256, 0, stream>>>(Wb1, wb1b, 128 * 3 * HDIM);

    // h0 = feat @ W0^T + b0
    k_gemm<<<mblocks, 256, 0, stream>>>(feat, nullptr, nullptr, 1,
                                        w0b, b0, HDIM,
                                        nullptr, nullptr, nullptr,
                                        h0, 0, N);

    // layer 1: agg(h0) ; h1 = agg*norm + norm2*([feat|h0] @ Wb0^T + bb0)
    (void)hipMemsetAsync(agg, 0, (size_t)N * HDIM * sizeof(float), stream);
    k_spmm<<<spmmB, 256, 0, stream>>>(h0, norm, src, dst, agg, E);
    k_gemm<<<mblocks, 256, 0, stream>>>(feat, h0, nullptr, 2,
                                        wb0b, bb0, 2 * HDIM,
                                        agg, norm, norm2,
                                        h1, 1, N);

    // layer 2: agg(h1) ; out = agg*norm + norm2*([feat|h0|h1] @ Wb1^T + bb1)
    (void)hipMemsetAsync(agg, 0, (size_t)N * HDIM * sizeof(float), stream);
    k_spmm<<<spmmB, 256, 0, stream>>>(h1, norm, src, dst, agg, E);
    k_gemm<<<mblocks, 256, 0, stream>>>(feat, h0, h1, 3,
                                        wb1b, bb1, 3 * HDIM,
                                        agg, norm, norm2,
                                        outp, 1, N);
}
